// SE3Transformer_20031727469171
// MI455X (gfx1250) — matrix operand order repaired and checked by reading, not compiled, not run on hardware
//
#include <hip/hip_runtime.h>
#include <math.h>

// ---------------------------------------------------------------------------
// SE(3)-Transformer forward for gfx1250 (MI455X).
// WMMA (v_wmma_f32_16x16x32_f16) is used for the radial-weight GEMMs, which
// dominate the FLOP count; irregular graph parts run on VALU + atomics.
// B matrix is packed in the exact WMMA lane layout so each fragment is one
// contiguous 32B load; A fragments are reused across 4 column tiles per wave.
// ---------------------------------------------------------------------------

#define RMAXC 0.0914f
#define WS_ECH 2048          // edge chunk for radial-weight scratch
#define WS_MAXNT 2048        // max padded concatenated path-weight columns
#define MAXP 64              // max paths packed per layer (k+v <= 58)
#define MAXPATH 36           // max paths per message kernel (<=34)
#define COLT 4               // column tiles per wave in the WMMA GEMM

typedef __attribute__((ext_vector_type(16))) _Float16 v16h;
typedef __attribute__((ext_vector_type(8)))  _Float16 half8;
typedef __attribute__((ext_vector_type(8)))  float    v8f;

static inline int cdiv(int a, int b) { return (a + b - 1) / b; }
static inline int imin(int a, int b) { return a < b ? a : b; }

// ------------------------------- kernels -----------------------------------

__global__ void k_fill_f32(float* p, int n, float v) {
  int i = blockIdx.x * blockDim.x + threadIdx.x;
  if (i < n) p[i] = v;
}

__global__ void k_fill_u32(unsigned* p, int n, unsigned v) {
  int i = blockIdx.x * blockDim.x + threadIdx.x;
  if (i < n) p[i] = v;
}

__global__ void k_copy_f32(float* d, const float* s, int n) {
  int i = blockIdx.x * blockDim.x + threadIdx.x;
  if (i < n) d[i] = s[i];
}

// Edge geometry: direction, real spherical harmonics (l<=3), radial basis (f16, K padded to 32)
__global__ void k_edge_geom(const float* pos, const int* src, const int* dst, int E,
                            float* sh, _Float16* basis16) {
  int e = blockIdx.x * blockDim.x + threadIdx.x;
  if (e >= E) return;
  int s = src[e], d = dst[e];
  float vx = pos[d * 3 + 0] - pos[s * 3 + 0];
  float vy = pos[d * 3 + 1] - pos[s * 3 + 1];
  float vz = pos[d * 3 + 2] - pos[s * 3 + 2];
  float r = sqrtf(vx * vx + vy * vy + vz * vz);
  float inv = 1.0f / (r + 1e-9f);
  float x = vx * inv, y = vy * inv, z = vz * inv;
  float* o = sh + (size_t)e * 16;
  const float s3 = 1.7320508f;
  o[0] = 1.0f;
  o[1] = x; o[2] = y; o[3] = z;
  o[4] = s3 * x * y; o[5] = s3 * y * z; o[6] = 0.5f * (3.f * z * z - 1.f);
  o[7] = s3 * z * x; o[8] = 0.5f * s3 * (x * x - y * y);
  float l3[7] = { y * (3.f * x * x - y * y), x * y * z, y * (5.f * z * z - 1.f),
                  z * (5.f * z * z - 3.f), x * (5.f * z * z - 1.f),
                  z * (x * x - y * y), x * (x * x - 3.f * y * y) };
  float n3 = 0.f;
  for (int i = 0; i < 7; ++i) n3 += l3[i] * l3[i];
  float i3 = 1.0f / (sqrtf(n3) + 1e-9f);
  for (int i = 0; i < 7; ++i) o[9 + i] = l3[i] * i3;
  _Float16* b = basis16 + (size_t)e * 32;
  for (int i = 0; i < 10; ++i) {
    float c = RMAXC * (float)i / 9.0f;                  // linspace(0, R_MAX, 10)
    float t = (r - c) / (RMAXC / 10.0f);
    b[i] = (_Float16)expf(-t * t);
  }
  for (int i = 10; i < 32; ++i) b[i] = (_Float16)0.0f;
}

// Pack per-path radial MLP weights rw[10, mi*mo] (f32) into the WMMA B-fragment
// layout: Bsw[tile][lane][e] where lane = 16*((K>>3)&1) + col%16 and e = (K&7) + 8*(K>=16).
// Each path's column block is padded to a multiple of 16; K rows 10..31 are zero.
// A lane's 16 B elements are then contiguous (one 32-byte load).
struct PackArgs {
  int np, NT;
  const float* rw[MAXP];
  int coff[MAXP];
  int sz[MAXP];
};

__global__ void k_pack_rw(PackArgs a, _Float16* Bsw) {
  int p = blockIdx.x;
  if (p >= a.np) return;
  int sz = a.sz[p];
  int pw = (sz + 15) & ~15;
  const float* w = a.rw[p];
  for (int t = threadIdx.x; t < 32 * pw; t += blockDim.x) {
    int kk = t / pw, j = t % pw;        // kk = K row (0..31), j = local column
    float v = (kk < 10 && j < sz) ? w[kk * sz + j] : 0.0f;
    int col = a.coff[p] + j;
    int tile = col >> 4, cl = col & 15;
    int g = (kk >> 3) & 1, e = (kk & 7) | ((kk >> 4) << 3);
    Bsw[(size_t)(tile * 32 + g * 16 + cl) * 16 + e] = (_Float16)v;
  }
}

// WMMA GEMM: W[e-e0, 0:NT] = basis16[e, 0:32] @ B[0:32, 0:NT]
// Each wave owns one 16-edge tile, loads A once, and sweeps COLT column tiles
// (one v_wmma_f32_16x16x32_f16 each) with a uniform fast path for full tiles.
// A layout (ISA 7.12.2, 16-bit A 16x32): lane L holds row M=L&15; lane half g=L>>4;
//   element e<8 -> K = 8g + e ; element e>=8 -> K = 16 + 8g + (e-8).
// C layout (16x16 f32): VGPR r, lane L -> M = r + 8g, N = L&15.
__global__ __launch_bounds__(128) void k_radial_gemm(const _Float16* basis16,
                                                     const _Float16* Bsw, float* W,
                                                     int e0, int e1, int NT) {
  int wave = threadIdx.x >> 5;
  int lane = threadIdx.x & 31;
  int row = lane & 15, g = lane >> 4;
  int tbase = e0 + (blockIdx.x * 4 + wave) * 16;   // this wave's 16-edge tile
  if (tbase >= e1) return;
  bool full = (tbase + 16 <= e1);
  v16h a;
  int edge = tbase + row;
  if (edge < e1) {
    const _Float16* base = basis16 + (size_t)edge * 32;
    half8 lo = *(const half8*)(base + 8 * g);
    half8 hi = *(const half8*)(base + 16 + 8 * g);
#pragma unroll
    for (int i = 0; i < 8; ++i) { a[i] = lo[i]; a[8 + i] = hi[i]; }
  } else {
#pragma unroll
    for (int i = 0; i < 16; ++i) a[i] = (_Float16)0.0f;
  }
  int ntiles = NT >> 4;
#pragma unroll
  for (int s = 0; s < COLT; ++s) {
    int tile = blockIdx.y * COLT + s;
    if (tile >= ntiles) break;
    v16h b = *(const v16h*)(Bsw + (size_t)(tile * 32 + lane) * 16);
    v8f c = {};
    c = __builtin_amdgcn_wmma_f32_16x16x32_f16(false, a, false, b, (short)0, c, false, false);
    float* wp = W + (size_t)(tbase - e0) * NT + tile * 16 + (lane & 15);
    if (full) {
#pragma unroll
      for (int r = 0; r < 8; ++r)
        wp[(size_t)(r + 8 * g) * NT] = c[r];
    } else {
#pragma unroll
      for (int r = 0; r < 8; ++r) {
        int er = tbase + r + 8 * g;
        if (er < e1) wp[(size_t)(r + 8 * g) * NT] = c[r];
      }
    }
  }
}

// Per-edge equivariant tensor product with path weights:
// out[e,k,o] = sum_paths sum_{m,i,j} x[src[e],m,i] * sh[e,j] * cg[i,j,o] * w[e,m,k]
struct MsgArgs {
  int np, e0, e1, NT, dimIn, dimOut;
  struct P {
    int mi, mo, di, df, dof, xoff, ooff, shoff, wcoff;
    const float* cg;
  } p[MAXPATH];
};

__global__ void k_message(MsgArgs A, const int* src, const float* xnode,
                          const float* sh, const float* Wrad, float* out) {
  int e = A.e0 + blockIdx.x * blockDim.x + threadIdx.x;
  if (e >= A.e1) return;
  const float* xe = xnode + (size_t)src[e] * A.dimIn;
  const float* se = sh + (size_t)e * 16;
  const float* we = Wrad + (size_t)(e - A.e0) * A.NT;
  float acc[80];
  for (int i = 0; i < A.dimOut; ++i) acc[i] = 0.0f;
  for (int pi = 0; pi < A.np; ++pi) {
    const MsgArgs::P& P = A.p[pi];
    float t[7][7];
    for (int i = 0; i < P.di; ++i)
      for (int o = 0; o < P.dof; ++o) {
        float s = 0.f;
        for (int j = 0; j < P.df; ++j) s += se[P.shoff + j] * P.cg[(i * P.df + j) * P.dof + o];
        t[i][o] = s;
      }
    float y[8][7];
    for (int m = 0; m < P.mi; ++m)
      for (int o = 0; o < P.dof; ++o) {
        float s = 0.f;
        for (int i = 0; i < P.di; ++i) s += xe[P.xoff + m * P.di + i] * t[i][o];
        y[m][o] = s;
      }
    const float* w = we + P.wcoff;
    for (int k = 0; k < P.mo; ++k)
      for (int o = 0; o < P.dof; ++o) {
        float s = 0.f;
        for (int m = 0; m < P.mi; ++m) s += w[m * P.mo + k] * y[m][o];
        acc[P.ooff + k * P.dof + o] += s;
      }
  }
  float* oe = out + (size_t)e * A.dimOut;
  for (int i = 0; i < A.dimOut; ++i) oe[i] = acc[i];
}

// Block-diagonal irrep linear: out[n, ooff + k*dc + c] (+)= sum_m in[n, ioff + m*dc + c] * W[m,k]
__global__ void k_linear(const float* in, const float* W, float* out,
                         int N, int ID, int OD, int ioff, int ooff,
                         int mi, int mo, int dc, int accum) {
  int idx = blockIdx.x * blockDim.x + threadIdx.x;
  int total = N * mo * dc;
  if (idx >= total) return;
  int n = idx / (mo * dc);
  int r = idx % (mo * dc);
  int k = r / dc, c = r % dc;
  float s = 0.f;
  for (int m = 0; m < mi; ++m) s += in[(size_t)n * ID + ioff + m * dc + c] * W[m * mo + k];
  float* o = out + (size_t)n * OD + ooff + k * dc + c;
  if (accum) *o += s; else *o = s;
}

__device__ __forceinline__ unsigned f32_key(float f) {
  unsigned u = __float_as_uint(f);
  return (u & 0x80000000u) ? ~u : (u | 0x80000000u);
}
__device__ __forceinline__ float key_f32(unsigned key) {
  unsigned u = (key & 0x80000000u) ? (key & 0x7FFFFFFFu) : ~key;
  return __uint_as_float(u);
}

__global__ void k_score(const float* q, const float* kbuf, const int* dst, int E,
                        float* score, unsigned* maxkey) {
  int e = blockIdx.x * blockDim.x + threadIdx.x;
  if (e >= E) return;
  const float* qe = q + (size_t)dst[e] * 36;
  const float* ke = kbuf + (size_t)e * 36;
  float s = 0.f;
  for (int i = 0; i < 36; ++i) s += qe[i] * ke[i];
  s *= (1.0f / 6.0f);                       // / sqrt(DIM_Q=36)
  score[e] = s;
  atomicMax(maxkey + dst[e], f32_key(s));
}

__global__ void k_exp(const float* score, const int* dst, int E, const unsigned* maxkey,
                      float* ex, float* den) {
  int e = blockIdx.x * blockDim.x + threadIdx.x;
  if (e >= E) return;
  float mx = key_f32(maxkey[dst[e]]);
  if (!isfinite(mx)) mx = 0.0f;
  float v = expf(score[e] - mx);
  ex[e] = v;
  atomicAdd(den + dst[e], v);
}

__global__ void k_agg(const float* ex, const float* den, const int* dst, int E, int dimO,
                      const float* vbuf, float* agg) {
  int idx = blockIdx.x * blockDim.x + threadIdx.x;
  int total = E * dimO;
  if (idx >= total) return;
  int e = idx / dimO, d = idx % dimO;
  float alpha = ex[e] / (den[dst[e]] + 1e-9f);
  atomicAdd(agg + (size_t)dst[e] * dimO + d, alpha * vbuf[(size_t)e * dimO + d]);
}

struct BlkArgs { int nb; int m[4]; int l[4]; int off[4]; int D; int doSkip; };

// norm-act: l=0 -> silu; l>0 -> b * silu(|b|)/(|b|+1e-9)
__global__ void k_normact(BlkArgs A, float* h, int N) {
  int idx = blockIdx.x * blockDim.x + threadIdx.x;
  int total = N * A.nb;
  if (idx >= total) return;
  int n = idx / A.nb, b = idx % A.nb;
  float* base = h + (size_t)n * A.D + A.off[b];
  int l = A.l[b], d = 2 * l + 1, m = A.m[b];
  if (l == 0) {
    for (int mm = 0; mm < m; ++mm) {
      float v = base[mm];
      base[mm] = v / (1.0f + expf(-v));
    }
  } else {
    for (int mm = 0; mm < m; ++mm) {
      float n2 = 0.f;
      for (int c = 0; c < d; ++c) { float v = base[mm * d + c]; n2 += v * v; }
      float nr = sqrtf(n2);
      float sc = (nr / (1.0f + expf(-nr))) / (nr + 1e-9f);
      for (int c = 0; c < d; ++c) base[mm * d + c] *= sc;
    }
  }
}

// optional skip-add + equivariant layernorm
__global__ void k_finish(BlkArgs A, float* y, const float* xold, int N) {
  int idx = blockIdx.x * blockDim.x + threadIdx.x;
  int total = N * A.nb;
  if (idx >= total) return;
  int n = idx / A.nb, b = idx % A.nb;
  int l = A.l[b], d = 2 * l + 1, m = A.m[b];
  float* base = y + (size_t)n * A.D + A.off[b];
  if (A.doSkip) {
    const float* xb = xold + (size_t)n * A.D + A.off[b];
    for (int i = 0; i < m * d; ++i) base[i] += xb[i];
  }
  if (l == 0) {
    float mean = 0.f;
    for (int mm = 0; mm < m; ++mm) mean += base[mm];
    mean /= (float)m;
    for (int mm = 0; mm < m; ++mm) base[mm] -= mean;
  }
  float acc = 0.f;
  for (int mm = 0; mm < m; ++mm) {
    float s = 0.f;
    for (int c = 0; c < d; ++c) { float v = base[mm * d + c]; s += v * v; }
    acc += s;
  }
  float rms = sqrtf(acc / (float)m / (float)d + 1e-6f);
  float ir = 1.0f / rms;
  for (int i = 0; i < m * d; ++i) base[i] *= ir;
}

// ------------------------------ host side ----------------------------------

struct Irr { int n; int m[4]; int l[4]; };

static int irr_dim(const Irr& a) { int d = 0; for (int i = 0; i < a.n; ++i) d += a.m[i] * (2 * a.l[i] + 1); return d; }
static int irr_off(const Irr& a, int b) { int d = 0; for (int i = 0; i < b; ++i) d += a.m[i] * (2 * a.l[i] + 1); return d; }
static int irr_find(const Irr& a, int l) { for (int i = 0; i < a.n; ++i) if (a.l[i] == l) return i; return -1; }

struct PB { int bi, lf, bo; };

static int build_paths(const Irr& A, const Irr& B, PB* out) {
  int n = 0;
  for (int bi = 0; bi < A.n; ++bi)
    for (int lf = 0; lf <= 3; ++lf)
      for (int bo = 0; bo < B.n; ++bo) {
        int li = A.l[bi], lo = B.l[bo];
        int dmin = li - lf; if (dmin < 0) dmin = -dmin;
        if (dmin <= lo && lo <= li + lf) { out[n].bi = bi; out[n].lf = lf; out[n].bo = bo; ++n; }
      }
  return n;
}

extern "C" void kernel_launch(void* const* d_in, const int* in_sizes, int n_in,
                              void* d_out, int out_size, void* d_ws, size_t ws_size,
                              hipStream_t stream) {
  const float* d_x  = (const float*)d_in[0];
  const float* d_pos = (const float*)d_in[1];
  const int* d_src = (const int*)d_in[2];
  const int* d_dst = (const int*)d_in[3];
  const int E = in_sizes[2];
  const int N = in_sizes[1] / 3;

  const Irr IRR_IN  = {1, {3, 0, 0, 0},  {0, 0, 0, 0}};
  const Irr IRR_EMB = {4, {8, 8, 4, 4},  {0, 1, 2, 3}};
  const Irr IRR_Q   = {3, {4, 4, 4, 0},  {0, 1, 2, 0}};
  const Irr IRR_LIN = {4, {16, 16, 8, 8}, {0, 1, 2, 3}};
  const Irr IRR_OUT = {2, {4, 4, 0, 0},  {0, 1, 0, 0}};

  struct LayerM {
    Irr in, out; int skipAdd;
    const float* Wl1[4]; const float* Wl2[4]; const float* Wq[4]; const float* Wsi[4];
    int nkp, nvp;
    PB kp[24], vp[34];
    const float* kcg[24]; const float* krw[24];
    const float* vcg[34]; const float* vrw[34];
    int kcoff[24], vcoff[34], NTtot;
  };
  static LayerM Ls[7];

  for (int li = 0; li < 7; ++li) {
    Ls[li].in  = (li == 0) ? IRR_IN : IRR_EMB;
    Ls[li].out = (li == 6) ? IRR_OUT : IRR_EMB;
    Ls[li].skipAdd = (li >= 1 && li <= 5) ? 1 : 0;  // skip && irr_x == irr_o
  }

  // Walk the jax-flattened param leaves (dict keys sorted: Wl1,Wl2,Wq,Wsi,k_cg,k_rw,v_cg,v_rw;
  // None leaves dropped).
  int idx = 4;
  for (int li = 0; li < 7; ++li) {
    LayerM& L = Ls[li];
    for (int b = 0; b < IRR_LIN.n; ++b)
      L.Wl1[b] = (irr_find(L.out, IRR_LIN.l[b]) >= 0) ? (const float*)d_in[idx++] : nullptr;
    for (int b = 0; b < L.out.n; ++b)
      L.Wl2[b] = (irr_find(IRR_LIN, L.out.l[b]) >= 0) ? (const float*)d_in[idx++] : nullptr;
    for (int b = 0; b < IRR_Q.n; ++b)
      L.Wq[b] = (irr_find(L.in, IRR_Q.l[b]) >= 0) ? (const float*)d_in[idx++] : nullptr;
    for (int b = 0; b < L.out.n; ++b)
      L.Wsi[b] = (irr_find(L.in, L.out.l[b]) >= 0) ? (const float*)d_in[idx++] : nullptr;
    L.nkp = build_paths(L.in, IRR_Q, L.kp);
    for (int p = 0; p < L.nkp; ++p) L.kcg[p] = (const float*)d_in[idx++];
    for (int p = 0; p < L.nkp; ++p) L.krw[p] = (const float*)d_in[idx++];
    L.nvp = build_paths(L.in, L.out, L.vp);
    for (int p = 0; p < L.nvp; ++p) L.vcg[p] = (const float*)d_in[idx++];
    for (int p = 0; p < L.nvp; ++p) L.vrw[p] = (const float*)d_in[idx++];
    // padded column offsets: k paths then v paths
    int wc = 0;
    for (int p = 0; p < L.nkp; ++p) {
      int sz = L.in.m[L.kp[p].bi] * IRR_Q.m[L.kp[p].bo];
      L.kcoff[p] = wc; wc += (sz + 15) & ~15;
    }
    for (int p = 0; p < L.nvp; ++p) {
      int sz = L.in.m[L.vp[p].bi] * L.out.m[L.vp[p].bo];
      L.vcoff[p] = wc; wc += (sz + 15) & ~15;
    }
    L.NTtot = wc;
  }

  // workspace carve-up
  size_t off = 0;
  auto carve = [&](size_t bytes) { size_t o = off; off = (off + bytes + 255) & ~(size_t)255; return o; };
  char* ws = (char*)d_ws;
  float*     shb  = (float*)(ws + carve((size_t)E * 16 * 4));
  _Float16*  b16  = (_Float16*)(ws + carve((size_t)E * 32 * 2));
  float*     xA   = (float*)(ws + carve((size_t)N * 80 * 4));
  float*     xB   = (float*)(ws + carve((size_t)N * 80 * 4));
  float*     qb   = (float*)(ws + carve((size_t)N * 36 * 4));
  float*     kb   = (float*)(ws + carve((size_t)E * 36 * 4));
  float*     vb   = (float*)(ws + carve((size_t)E * 80 * 4));
  float*     sc   = (float*)(ws + carve((size_t)E * 4));
  float*     exb  = (float*)(ws + carve((size_t)E * 4));
  unsigned*  mk   = (unsigned*)(ws + carve((size_t)N * 4));
  float*     den  = (float*)(ws + carve((size_t)N * 4));
  float*     agg  = (float*)(ws + carve((size_t)N * 80 * 4));
  float*     hb   = (float*)(ws + carve((size_t)N * 160 * 4));
  _Float16*  Bp   = (_Float16*)(ws + carve((size_t)32 * WS_MAXNT * 2));
  float*     Wr   = (float*)(ws + carve((size_t)WS_ECH * WS_MAXNT * 4));
  (void)ws_size; (void)n_in;

  const int T = 256;
  auto lin_launch = [&](const float* in, float* out, const Irr& ia, const Irr& ib,
                        const float* const* Ws, int accum) {
    int ID = irr_dim(ia), OD = irr_dim(ib);
    for (int b = 0; b < ib.n; ++b) {
      if (!Ws[b]) continue;
      int bi = irr_find(ia, ib.l[b]);
      int mi = ia.m[bi], mo = ib.m[b], dc = 2 * ib.l[b] + 1;
      int total = N * mo * dc;
      k_linear<<<cdiv(total, T), T, 0, stream>>>(in, Ws[b], out, N, ID, OD,
                                                 irr_off(ia, bi), irr_off(ib, b),
                                                 mi, mo, dc, accum);
    }
  };

  // edge geometry + f16 basis (once)
  if (E > 0)
    k_edge_geom<<<cdiv(E, T), T, 0, stream>>>(d_pos, d_src, d_dst, E, shb, b16);

  // x -> xA (layer-0 feature layout is contiguous N*3)
  k_copy_f32<<<cdiv(N * 3, T), T, 0, stream>>>(xA, d_x, N * 3);

  float* cur = xA;
  float* nxt = xB;

  for (int li = 0; li < 7; ++li) {
    LayerM& L = Ls[li];
    const int dimIn = irr_dim(L.in), dimOut = irr_dim(L.out);

    // zero scratch used this layer
    k_fill_f32<<<cdiv(N * 36, T), T, 0, stream>>>(qb, N * 36, 0.f);
    k_fill_u32<<<cdiv(N, T), T, 0, stream>>>(mk, N, 0u);
    k_fill_f32<<<cdiv(N, T), T, 0, stream>>>(den, N, 0.f);
    k_fill_f32<<<cdiv(N * dimOut, T), T, 0, stream>>>(agg, N * dimOut, 0.f);
    k_fill_f32<<<cdiv(N * 160, T), T, 0, stream>>>(hb, N * 160, 0.f);

    // q = irrep_linear(x)
    lin_launch(cur, qb, L.in, IRR_Q, L.Wq, 0);

    // pack all radial weights (k paths then v paths) into WMMA-swizzled f16 B
    PackArgs PA;
    PA.np = L.nkp + L.nvp;
    PA.NT = L.NTtot;
    for (int p = 0; p < L.nkp; ++p) {
      PA.rw[p] = L.krw[p]; PA.coff[p] = L.kcoff[p];
      PA.sz[p] = L.in.m[L.kp[p].bi] * IRR_Q.m[L.kp[p].bo];
    }
    for (int p = 0; p < L.nvp; ++p) {
      PA.rw[L.nkp + p] = L.vrw[p]; PA.coff[L.nkp + p] = L.vcoff[p];
      PA.sz[L.nkp + p] = L.in.m[L.vp[p].bi] * L.out.m[L.vp[p].bo];
    }
    k_pack_rw<<<PA.np, 128, 0, stream>>>(PA, Bp);

    // message metadata
    MsgArgs Mk, Mv;
    auto fill_msg = [&](MsgArgs& M, bool isK) {
      const Irr& IA = L.in;
      const Irr& IO = isK ? IRR_Q : L.out;
      M.np = isK ? L.nkp : L.nvp;
      M.NT = L.NTtot;
      M.dimIn = dimIn;
      M.dimOut = irr_dim(IO);
      for (int p = 0; p < M.np; ++p) {
        const PB& pb = isK ? L.kp[p] : L.vp[p];
        MsgArgs::P& P = M.p[p];
        P.mi = IA.m[pb.bi]; P.mo = IO.m[pb.bo];
        P.di = 2 * IA.l[pb.bi] + 1; P.df = 2 * pb.lf + 1; P.dof = 2 * IO.l[pb.bo] + 1;
        P.xoff = irr_off(IA, pb.bi); P.ooff = irr_off(IO, pb.bo);
        P.shoff = pb.lf * pb.lf;
        P.wcoff = isK ? L.kcoff[p] : L.vcoff[p];
        P.cg = isK ? L.kcg[p] : L.vcg[p];
      }
    };
    fill_msg(Mk, true);
    fill_msg(Mv, false);

    // chunked: WMMA radial GEMM then per-edge tensor-product messages (k and v)
    for (int e0 = 0; e0 < E; e0 += WS_ECH) {
      int e1 = imin(e0 + WS_ECH, E);
      int ech = e1 - e0;
      dim3 gg(cdiv(ech, 64), cdiv(L.NTtot / 16, COLT));
      k_radial_gemm<<<gg, 128, 0, stream>>>(b16, Bp, Wr, e0, e1, L.NTtot);
      Mk.e0 = e0; Mk.e1 = e1;
      k_message<<<cdiv(ech, T), T, 0, stream>>>(Mk, d_src, cur, shb, Wr, kb);
      Mv.e0 = e0; Mv.e1 = e1;
      k_message<<<cdiv(ech, T), T, 0, stream>>>(Mv, d_src, cur, shb, Wr, vb);
    }

    // attention: score, segment max, exp/sum, weighted aggregation
    if (E > 0) {
      k_score<<<cdiv(E, T), T, 0, stream>>>(qb, kb, d_dst, E, sc, mk);
      k_exp<<<cdiv(E, T), T, 0, stream>>>(sc, d_dst, E, mk, exb, den);
      k_agg<<<cdiv(E * dimOut, T), T, 0, stream>>>(exb, den, d_dst, E, dimOut, vb, agg);
    }

    // h = Wl1(agg); norm_act; y = Wl2(h) + Wsi(x) (+x); eq_layernorm
    lin_launch(agg, hb, L.out, IRR_LIN, L.Wl1, 0);

    BlkArgs AL;
    AL.nb = IRR_LIN.n; AL.D = 160; AL.doSkip = 0;
    for (int b = 0; b < IRR_LIN.n; ++b) { AL.m[b] = IRR_LIN.m[b]; AL.l[b] = IRR_LIN.l[b]; AL.off[b] = irr_off(IRR_LIN, b); }
    k_normact<<<cdiv(N * AL.nb, T), T, 0, stream>>>(AL, hb, N);

    lin_launch(hb, nxt, IRR_LIN, L.out, L.Wl2, 0);   // Wl2 present for all out blocks
    lin_launch(cur, nxt, L.in, L.out, L.Wsi, 1);     // self-interaction accumulate

    BlkArgs AO;
    AO.nb = L.out.n; AO.D = dimOut; AO.doSkip = L.skipAdd;
    for (int b = 0; b < L.out.n; ++b) { AO.m[b] = L.out.m[b]; AO.l[b] = L.out.l[b]; AO.off[b] = irr_off(L.out, b); }
    k_finish<<<cdiv(N * AO.nb, T), T, 0, stream>>>(AO, nxt, cur, N);

    float* t = cur; cur = nxt; nxt = t;
  }

  // final features (IRR_OUT: dim 16, contiguous) -> d_out
  k_copy_f32<<<cdiv(out_size, T), T, 0, stream>>>((float*)d_out, cur, out_size);
}
